// TiedEmbeddingLinear_77068893159750
// MI455X (gfx1250) — matrix operand order repaired and checked by reading, not compiled, not run on hardware
//
#include <hip/hip_runtime.h>
#include <hip/hip_bf16.h>
#include <stdint.h>

// Problem constants (from reference)
#define VOCAB   32768
#define DMODEL  4096
#define PACKEDW 2048          // DMODEL/2, one byte (2 nibbles) per int32
#define MTOT    8192          // BATCH*SEQ
#define KTOT    DMODEL
#define NTOT    VOCAB

typedef __attribute__((ext_vector_type(16))) __bf16 v16bf;
typedef __attribute__((ext_vector_type(8)))  __bf16 v8bf;
typedef __attribute__((ext_vector_type(8)))  float  v8f;

union Frag { uint4 u[2]; v16bf v; };

// ---------------------------------------------------------------------------
// Phase 1a: NF4 -> bf16 dequant of W. One thread handles 4 packed int32s
// (8 output elements, one 16B store). codebook kept in LDS for dynamic index.
// ---------------------------------------------------------------------------
__global__ __launch_bounds__(256)
void nf4_dequant_w(const int* __restrict__ q, const float* __restrict__ absmax,
                   const float* __restrict__ codebook, v8bf* __restrict__ wout) {
    __shared__ float cb[16];
    if (threadIdx.x < 16) cb[threadIdx.x] = codebook[threadIdx.x];
    __syncthreads();
    size_t idx = (size_t)blockIdx.x * 256 + threadIdx.x;   // 16,777,216 total
    int row = (int)(idx >> 9);                             // 512 int4-groups per vocab row
    float amax = absmax[row];
    int4 qq = reinterpret_cast<const int4*>(q)[idx];
    int vals[4] = {qq.x, qq.y, qq.z, qq.w};
    v8bf o;
#pragma unroll
    for (int i = 0; i < 4; ++i) {
        int hi = (vals[i] >> 4) & 15;
        int lo =  vals[i]       & 15;
        o[2*i]   = (__bf16)(cb[hi] * amax);
        o[2*i+1] = (__bf16)(cb[lo] * amax);
    }
    wout[idx] = o;
}

// ---------------------------------------------------------------------------
// Phase 1b: x fp32 -> bf16 (67 MB result: fully L2-resident for the GEMM)
// ---------------------------------------------------------------------------
__global__ __launch_bounds__(256)
void cvt_x_bf16(const float4* __restrict__ x, v8bf* __restrict__ xout) {
    size_t idx = (size_t)blockIdx.x * 256 + threadIdx.x;   // 4,194,304 total (8 floats each)
    float4 a = x[2*idx], b = x[2*idx + 1];
    v8bf o;
    o[0] = (__bf16)a.x; o[1] = (__bf16)a.y; o[2] = (__bf16)a.z; o[3] = (__bf16)a.w;
    o[4] = (__bf16)b.x; o[5] = (__bf16)b.y; o[6] = (__bf16)b.z; o[7] = (__bf16)b.w;
    xout[idx] = o;
}

// ---------------------------------------------------------------------------
// Phase 2: C[M,N] = A[M,K](bf16) * B[N,K](bf16)^T with v_wmma_f32_16x16x32_bf16
// Block tile 128x128, BK=64, double-buffered LDS filled with
// global_load_async_to_lds_b128, synchronized with s_wait_asynccnt.
// 8 waves: 2 (M) x 4 (N); each wave computes 4x2 16x16 fp32 accum tiles.
// ---------------------------------------------------------------------------
#define LDT_B   144            // padded LDS row pitch in bytes (64*2 + 16)
#define STAGE_A 18432          // 128 * 144
#define STAGE_T 36864          // A + B per stage
#define NSTAGE  (KTOT/64)      // 64

__global__ __launch_bounds__(256)
void gemm_bf16_wmma(const __bf16* __restrict__ A, const __bf16* __restrict__ B,
                    float* __restrict__ C) {
    __shared__ __align__(16) unsigned char lds[2 * STAGE_T];   // 73,728 B

    const int tid  = threadIdx.x;
    const int lane = tid & 31;
    const int wave = tid >> 5;
    const int wm   = wave >> 2;        // 0..1 -> 64 M rows
    const int wn   = wave & 3;         // 0..3 -> 32 N cols
    const int m0   = blockIdx.x * 128;
    const int n0   = blockIdx.y * 128;

    const uint32_t ldsBase = (uint32_t)(uintptr_t)lds;

    // Per-thread copy plan: 4 chunks of 16B for A and for B per stage.
    uint32_t aOff[4], bOff[4], lOff[4];
#pragma unroll
    for (int i = 0; i < 4; ++i) {
        int c = tid + 256 * i;         // 1024 chunks per tile
        int row = c >> 3;              // 8 chunks (128B) per 64-elem bf16 row
        int col = c & 7;
        aOff[i] = ((uint32_t)(m0 + row) * KTOT + col * 8) * 2;
        bOff[i] = ((uint32_t)(n0 + row) * KTOT + col * 8) * 2;
        lOff[i] = (uint32_t)(row * LDT_B + col * 16);
    }

    auto issue_stage = [&](int s, int buf) {
        uint32_t kbyte = (uint32_t)s * 128u;   // 64 elems * 2B
#pragma unroll
        for (int i = 0; i < 4; ++i) {
            uint32_t la = ldsBase + (uint32_t)buf * STAGE_T + lOff[i];
            uint32_t ga = aOff[i] + kbyte;
            asm volatile("global_load_async_to_lds_b128 %0, %1, %2"
                         :: "v"(la), "v"(ga), "s"(A) : "memory");
            uint32_t lb = la + STAGE_A;
            uint32_t gb = bOff[i] + kbyte;
            asm volatile("global_load_async_to_lds_b128 %0, %1, %2"
                         :: "v"(lb), "v"(gb), "s"(B) : "memory");
        }
    };

    v8f acc[4][2] = {};

    const int half = lane >> 4;        // A: K-half selector; B: K-half selector
    const int l16  = lane & 15;

    issue_stage(0, 0);
    for (int s = 0; s < NSTAGE; ++s) {
        const int buf = s & 1;
        if (s + 1 < NSTAGE) {
            issue_stage(s + 1, buf ^ 1);
            asm volatile("s_wait_asynccnt 0x8" ::: "memory");  // stage s complete
        } else {
            asm volatile("s_wait_asynccnt 0x0" ::: "memory");
        }
        __syncthreads();

        const uint32_t aBase = (uint32_t)buf * STAGE_T;
        const uint32_t bBase = aBase + STAGE_A;
#pragma unroll
        for (int kk = 0; kk < 2; ++kk) {          // two K=32 steps per BK=64 stage
            Frag afr[4], bfr[2];
#pragma unroll
            for (int ms = 0; ms < 4; ++ms) {
                // A 16x32 layout: lanes0-15 K{0..7,16..23}; lanes16-31 K{8..15,24..31}
                int row = wm * 64 + ms * 16 + l16;
                uint32_t off = aBase + (uint32_t)row * LDT_B + (kk * 32 + half * 8) * 2;
                afr[ms].u[0] = *(const uint4*)&lds[off];
                afr[ms].u[1] = *(const uint4*)&lds[off + 32];
            }
#pragma unroll
            for (int ns = 0; ns < 2; ++ns) {
                // B 32x16 layout: lanes0-15 K{0..7,16..23}; lanes16-31 K{8..15,24..31}
                int row = wn * 32 + ns * 16 + l16;
                uint32_t off = bBase + (uint32_t)row * LDT_B + (kk * 32 + half * 8) * 2;
                bfr[ns].u[0] = *(const uint4*)&lds[off];
                bfr[ns].u[1] = *(const uint4*)&lds[off + 32];
            }
#pragma unroll
            for (int ms = 0; ms < 4; ++ms)
#pragma unroll
                for (int ns = 0; ns < 2; ++ns)
                    acc[ms][ns] = __builtin_amdgcn_wmma_f32_16x16x32_bf16(
                        false, afr[ms].v, false, bfr[ns].v,
                        (short)0, acc[ms][ns], false, false);
        }
        __syncthreads();
    }

    // C/D layout: VGPR r, lane l -> row = (l>=16)*8 + r, col = l&15
#pragma unroll
    for (int ms = 0; ms < 4; ++ms) {
        int mrow = m0 + wm * 64 + ms * 16 + half * 8;
#pragma unroll
        for (int ns = 0; ns < 2; ++ns) {
            int ncol = n0 + wn * 32 + ns * 16 + l16;
#pragma unroll
            for (int r = 0; r < 8; ++r)
                C[(size_t)(mrow + r) * NTOT + ncol] = acc[ms][ns][r];
        }
    }
}

// ---------------------------------------------------------------------------
extern "C" void kernel_launch(void* const* d_in, const int* in_sizes, int n_in,
                              void* d_out, int out_size, void* d_ws, size_t ws_size,
                              hipStream_t stream) {
    const float* x        = (const float*)d_in[0];
    const int*   q        = (const int*)  d_in[1];
    const float* absmax   = (const float*)d_in[2];
    const float* codebook = (const float*)d_in[3];
    float* out = (float*)d_out;

    __bf16* wBF = (__bf16*)d_ws;                                    // 256 MiB
    __bf16* xBF = (__bf16*)((char*)d_ws + (size_t)NTOT * KTOT * 2); // +64 MiB

    // Phase 1a: dequant W (16,777,216 int4-groups / 256)
    nf4_dequant_w<<<65536, 256, 0, stream>>>(q, absmax, codebook, (v8bf*)wBF);
    // Phase 1b: x -> bf16 (4,194,304 groups / 256)
    cvt_x_bf16<<<16384, 256, 0, stream>>>((const float4*)x, (v8bf*)xBF);
    // Phase 2: GEMM. M-major grid within each N column for L2 weight reuse.
    gemm_bf16_wmma<<<dim3(MTOT / 128, NTOT / 128), 256, 0, stream>>>(xBF, wBF, out);
}
